// InteractionLayer_29850022707551
// MI455X (gfx1250) — matrix operand order repaired and checked by reading, not compiled, not run on hardware
//
#include <hip/hip_runtime.h>

#define C_DIM 64
#define AGG_W 576   // 64 + 64*3 + 64*5

typedef __attribute__((ext_vector_type(16))) __bf16        v16bf;
typedef __attribute__((ext_vector_type(8)))  float         v8f;
typedef __attribute__((ext_vector_type(8)))  unsigned int  v8u;

__device__ __forceinline__ unsigned short f2bf(float f) {
  unsigned u = __builtin_bit_cast(unsigned, f);
  unsigned r = u + 0x7FFFu + ((u >> 16) & 1u);   // round-to-nearest-even
  return (unsigned short)(r >> 16);
}
__device__ __forceinline__ float bf2f(unsigned short s) {
  return __builtin_bit_cast(float, ((unsigned)s) << 16);
}
__device__ __forceinline__ void lds_fence_wave() {
  asm volatile("s_wait_dscnt 0" ::: "memory");   // wave-local LDS ordering (CDNA5 split counters)
}

// ---- WMMA bf16 16x16x32, f32 accumulate -----------------------------------
__device__ __forceinline__ v8f wmma_bf16(v16bf a, v16bf b, v8f c) {
  return __builtin_amdgcn_wmma_f32_16x16x32_bf16(false, a, false, b, (short)0, c, false, false);
}

// A fragment (16x32 bf16, row-major LDS tile, row stride in halfwords).
// lane m=lane&15 both halves; VGPR j pair base = 2*j + 16*half (contiguous, matches B).
__device__ __forceinline__ v16bf load_frag_A(const unsigned short* act, int strideH,
                                             int m, int half, int k0) {
  const unsigned int* row = (const unsigned int*)(act + m * strideH + k0);
  v8u w;
#pragma unroll
  for (int j = 0; j < 8; ++j) {
    int kb = (j * 2) + (half << 4);
    w[j] = row[kb >> 1];
  }
  return __builtin_bit_cast(v16bf, w);
}

// B fragment (32x16 bf16) from LDS weight stored TRANSPOSED: wT[n*Kw + k].
// lanes 0-15: N=lane, K=k0..k0+15; lanes 16-31: N=lane-16, K=k0+16..k0+31 -> 8 contiguous dwords.
__device__ __forceinline__ v16bf load_frag_B(const unsigned short* wT, int Kw,
                                             int n, int half, int k0) {
  const unsigned int* row = (const unsigned int*)(wT + n * Kw + k0 + (half << 4));
  v8u w;
#pragma unroll
  for (int j = 0; j < 8; ++j) w[j] = row[j];
  return __builtin_bit_cast(v16bf, w);
}

// One 16-row x (nTiles*16)-col bf16 GEMM layer out of LDS into LDS.
__device__ __forceinline__ void run_layer(const unsigned short* in, int inStrideH, int kChunks,
                                          const unsigned short* wT, int Kw,
                                          int nTiles, int nColBase,
                                          unsigned short* outBuf, int outStrideH,
                                          float scale, bool do_silu, int m, int half) {
#pragma unroll
  for (int nt = 0; nt < 4; ++nt) {
    if (nt >= nTiles) break;
    v8f acc = {};
#pragma unroll
    for (int kc = 0; kc < 2; ++kc) {
      if (kc >= kChunks) break;
      v16bf a = load_frag_A(in, inStrideH, m, half, kc * 32);
      v16bf b = load_frag_B(wT, Kw, nColBase + nt * 16 + m, half, kc * 32);
      acc = wmma_bf16(a, b, acc);
    }
#pragma unroll
    for (int j = 0; j < 8; ++j) {
      float x = acc[j] * scale;
      if (do_silu) x = x / (1.0f + __expf(-x));
      outBuf[(j + 8 * half) * outStrideH + nt * 16 + m] = f2bf(x);
    }
  }
}

// ============================ Kernel 1: up-projection =======================
// h = node_feats @ w_up * (1/8)
__global__ __launch_bounds__(256) void k_up(const float* __restrict__ nf,
                                            const float* __restrict__ w,
                                            float* __restrict__ h, int nTiles) {
  __shared__ unsigned short sW[64 * 64];
  __shared__ unsigned short sA[8][16 * 64];
  int tid = threadIdx.x;
  for (int i = tid; i < 64 * 64; i += 256) {
    int k = i >> 6, n = i & 63;
    sW[n * 64 + k] = f2bf(w[i]);
  }
  __syncthreads();
  int wave = tid >> 5, lane = tid & 31;
  int tile = blockIdx.x * 8 + wave;
  if (tile >= nTiles) return;
  unsigned short* A = sA[wave];
  int row0 = tile * 16;
  for (int i = lane; i < 16 * 64; i += 32) A[i] = f2bf(nf[(size_t)row0 * 64 + i]);
  lds_fence_wave();
  int m = lane & 15, half = lane >> 4;
#pragma unroll
  for (int nt = 0; nt < 4; ++nt) {
    v8f acc = {};
#pragma unroll
    for (int kc = 0; kc < 2; ++kc) {
      v16bf a = load_frag_A(A, 64, m, half, kc * 32);
      v16bf b = load_frag_B(sW, 64, nt * 16 + m, half, kc * 32);
      acc = wmma_bf16(a, b, acc);
    }
#pragma unroll
    for (int j = 0; j < 8; ++j)
      h[(size_t)(row0 + j + 8 * half) * 64 + nt * 16 + m] = acc[j] * 0.125f;
  }
}

// ============================ Kernel 2: edge MLP + messages =================
__global__ __launch_bounds__(128) void k_edge(const float* __restrict__ vec,
                                              const float* __restrict__ rad,
                                              const int* __restrict__ snd,
                                              const int* __restrict__ rcv,
                                              const float* __restrict__ w1,
                                              const float* __restrict__ w2,
                                              const float* __restrict__ w3,
                                              const float* __restrict__ w4,
                                              const float* __restrict__ h,
                                              float* __restrict__ agg, int nTiles) {
  __shared__ unsigned short sW1[64 * 32];    // (8->64), K padded to 32
  __shared__ unsigned short sW2[64 * 64];
  __shared__ unsigned short sW3[64 * 64];
  __shared__ unsigned short sW4[192 * 64];
  __shared__ unsigned short sPing[4][16 * 64];
  __shared__ unsigned short sPong[4][16 * 64];
  __shared__ float          sY[4][16 * 8];
  __shared__ int            sSend[4][16];
  __shared__ int            sRecv[4][16];

  int tid = threadIdx.x;
  for (int i = tid; i < 64 * 32; i += 128) sW1[i] = 0;
  __syncthreads();
  for (int i = tid; i < 8 * 64; i += 128) { int k = i >> 6, n = i & 63; sW1[n * 32 + k] = f2bf(w1[i]); }
  for (int i = tid; i < 64 * 64; i += 128) {
    int k = i >> 6, n = i & 63;
    sW2[n * 64 + k] = f2bf(w2[i]);
    sW3[n * 64 + k] = f2bf(w3[i]);
  }
  for (int i = tid; i < 64 * 192; i += 128) {
    int k = i / 192, n = i - k * 192;
    sW4[n * 64 + k] = f2bf(w4[i]);
  }
  __syncthreads();

  int wave = tid >> 5, lane = tid & 31;
  int tile = blockIdx.x * 4 + wave;
  if (tile >= nTiles) return;
  int e0 = tile * 16;
  unsigned short* ping = sPing[wave];
  unsigned short* pong = sPong[wave];
  float* yb = sY[wave];

  if (lane < 16) {
    int e = e0 + lane;
    float vx = vec[3 * e + 0], vy = vec[3 * e + 1], vz = vec[3 * e + 2];
    float ri = rsqrtf(vx * vx + vy * vy + vz * vz);
    float x = vx * ri, y = vy * ri, z = vz * ri;
    const float s3 = 1.7320508075688772f;   // sqrt(3)
    const float s15 = 3.872983346207417f;   // sqrt(15)
    const float hs5 = 1.1180339887498949f;  // 0.5*sqrt(5)
    float* yy = &yb[lane * 8];
    yy[0] = s3 * x; yy[1] = s3 * y; yy[2] = s3 * z;
    yy[3] = s15 * x * y; yy[4] = s15 * y * z;
    yy[5] = hs5 * (3.0f * z * z - 1.0f);
    yy[6] = s15 * x * z; yy[7] = 0.5f * s15 * (x * x - y * y);
    sSend[wave][lane] = snd[e];
    sRecv[wave][lane] = rcv[e];
  }
  // radial -> ping[16][32], K-padded with zeros
  for (int i = lane; i < 16 * 32; i += 32) ping[i] = 0;
  for (int i = lane; i < 16 * 8; i += 32) {
    int r = i >> 3, k = i & 7;
    ping[r * 32 + k] = f2bf(rad[(size_t)e0 * 8 + i]);
  }
  lds_fence_wave();

  int m = lane & 15, half = lane >> 4;
  const float RS8 = 0.35355339059327373f;  // 1/sqrt(8)
  run_layer(ping, 32, 1, sW1, 32, 4, 0, pong, 64, RS8, true, m, half);
  lds_fence_wave();
  run_layer(pong, 64, 2, sW2, 64, 4, 0, ping, 64, 0.125f, true, m, half);
  lds_fence_wave();
  run_layer(ping, 64, 2, sW3, 64, 4, 0, pong, 64, 0.125f, true, m, half);
  lds_fence_wave();

  const float GSC = 0.125f * (1.0f / 64.0f);  // mlp_w4 1/sqrt(64) and msg 1/C
  for (int ci = 0; ci < 3; ++ci) {
    run_layer(pong, 64, 2, sW4, 64, 4, ci * 64, ping, 64, GSC, false, m, half);
    lds_fence_wave();
    for (int t = lane; t < 16 * 64; t += 32) {
      int e = t >> 6, c = t & 63;
      float g = bf2f(ping[t]);
      float sv = h[(size_t)sSend[wave][e] * 64 + c];
      float base = sv * g;
      float* ag = agg + (size_t)sRecv[wave][e] * AGG_W;
      if (ci == 0) {
        atomicAdd(ag + c, base);
      } else if (ci == 1) {
        float* yy = &yb[e * 8];
#pragma unroll
        for (int q = 0; q < 3; ++q) atomicAdd(ag + 64 + c * 3 + q, base * yy[q]);
      } else {
        float* yy = &yb[e * 8 + 3];
#pragma unroll
        for (int q = 0; q < 5; ++q) atomicAdd(ag + 256 + c * 5 + q, base * yy[q]);
      }
    }
    lds_fence_wave();
  }
}

// ============================ Kernel 3: down-projections ====================
// out slices = (agg*EPS) @ w_down{0,1,2} * (1/8)  -> combined scale 1/16
__global__ __launch_bounds__(256) void k_down(const float* __restrict__ agg,
                                              const float* __restrict__ wd0,
                                              const float* __restrict__ wd1,
                                              const float* __restrict__ wd2,
                                              float* __restrict__ out, int nNodeTiles) {
  __shared__ unsigned short sW[3][64 * 64];
  __shared__ unsigned short sA[8][16 * 64];
  int tid = threadIdx.x;
  for (int i = tid; i < 64 * 64; i += 256) {
    int k = i >> 6, n = i & 63;
    sW[0][n * 64 + k] = f2bf(wd0[i]);
    sW[1][n * 64 + k] = f2bf(wd1[i]);
    sW[2][n * 64 + k] = f2bf(wd2[i]);
  }
  __syncthreads();
  int wave = tid >> 5, lane = tid & 31;
  int task = blockIdx.x * 8 + wave;
  if (task >= nNodeTiles * 9) return;
  int tile = task / 9, s = task - tile * 9;
  int wsel, cbase, strd, moff;
  if (s == 0)      { wsel = 0; cbase = 0;   strd = 1; moff = 0; }
  else if (s < 4)  { wsel = 1; cbase = 64;  strd = 3; moff = s - 1; }
  else             { wsel = 2; cbase = 256; strd = 5; moff = s - 4; }

  unsigned short* A = sA[wave];
  int row0 = tile * 16;
  for (int i = lane; i < 16 * 64; i += 32) {
    int r = i >> 6, c = i & 63;
    A[i] = f2bf(agg[(size_t)(row0 + r) * AGG_W + cbase + c * strd + moff]);
  }
  lds_fence_wave();
  int m = lane & 15, half = lane >> 4;
#pragma unroll
  for (int nt = 0; nt < 4; ++nt) {
    v8f acc = {};
#pragma unroll
    for (int kc = 0; kc < 2; ++kc) {
      v16bf a = load_frag_A(A, 64, m, half, kc * 32);
      v16bf b = load_frag_B(sW[wsel], 64, nt * 16 + m, half, kc * 32);
      acc = wmma_bf16(a, b, acc);
    }
#pragma unroll
    for (int j = 0; j < 8; ++j) {
      int d = nt * 16 + m;
      out[(size_t)(row0 + j + 8 * half) * AGG_W + cbase + d * strd + moff] = acc[j] * 0.0625f;
    }
  }
}

// ============================ Launch =======================================
extern "C" void kernel_launch(void* const* d_in, const int* in_sizes, int n_in,
                              void* d_out, int out_size, void* d_ws, size_t ws_size,
                              hipStream_t stream) {
  const float* vectors    = (const float*)d_in[0];
  const float* node_feats = (const float*)d_in[1];
  const float* radial     = (const float*)d_in[2];
  const int*   senders    = (const int*)d_in[3];
  const int*   receivers  = (const int*)d_in[4];
  const float* w_up       = (const float*)d_in[5];
  const float* mlp_w1     = (const float*)d_in[6];
  const float* mlp_w2     = (const float*)d_in[7];
  const float* mlp_w3     = (const float*)d_in[8];
  const float* mlp_w4     = (const float*)d_in[9];
  const float* w_down0    = (const float*)d_in[10];
  const float* w_down1    = (const float*)d_in[11];
  const float* w_down2    = (const float*)d_in[12];
  float* out = (float*)d_out;

  int nEdges = in_sizes[3];
  int nNodes = in_sizes[1] / C_DIM;
  int nodeTiles = nNodes / 16;   // 1250
  int edgeTiles = nEdges / 16;   // 20000

  float* h   = (float*)d_ws;
  float* agg = (float*)((char*)d_ws + (size_t)nNodes * C_DIM * sizeof(float));

  hipMemsetAsync(agg, 0, (size_t)nNodes * AGG_W * sizeof(float), stream);

  k_up<<<(nodeTiles + 7) / 8, 256, 0, stream>>>(node_feats, w_up, h, nodeTiles);
  k_edge<<<(edgeTiles + 3) / 4, 128, 0, stream>>>(vectors, radial, senders, receivers,
                                                  mlp_w1, mlp_w2, mlp_w3, mlp_w4,
                                                  h, agg, edgeTiles);
  k_down<<<(nodeTiles * 9 + 7) / 8, 256, 0, stream>>>(agg, w_down0, w_down1, w_down2,
                                                      out, nodeTiles);
}
